// MoE_47278999994656
// MI455X (gfx1250) — matrix operand order repaired and checked by reading, not compiled, not run on hardware
//
#include <hip/hip_runtime.h>
#include <hip/hip_bf16.h>

typedef __attribute__((ext_vector_type(16))) __bf16        v16bf;
typedef __attribute__((ext_vector_type(8)))  float         v8f;
typedef __attribute__((ext_vector_type(4)))  unsigned int  u32x4;

#define XPAD 136   // 128 + 8 bf16 pad -> 272B rows (16B aligned, de-phased banks)
#define HW   6400
#define CH   256
#define HALF 128
#define QTR  64
#define NEXP 5

__device__ __forceinline__ unsigned short f2bf(float x) {
  unsigned int u = __float_as_uint(x);
  u += 0x7FFFu + ((u >> 16) & 1u);          // round-to-nearest-even
  return (unsigned short)(u >> 16);
}
__device__ __forceinline__ float bf2f(unsigned short h) {
  return __uint_as_float(((unsigned int)h) << 16);
}
__device__ __forceinline__ v16bf ld_frag(const unsigned short* p0, const unsigned short* p1) {
  union { u32x4 u[2]; v16bf v; } f;
  f.u[0] = *reinterpret_cast<const u32x4*>(p0);
  f.u[1] = *reinterpret_cast<const u32x4*>(p1);
  return f.v;
}
// A fragment 16x32 bf16 from row-major LDS [M][XPAD]:
// lanes 0-15 hold K {0..7,16..23}, lanes 16-31 hold K {8..15,24..31}
__device__ __forceinline__ v16bf ld_a(const unsigned short* base, int row, int kt, int hi) {
  const unsigned short* p = base + row * XPAD + kt * 32 + hi * 8;
  return ld_frag(p, p + 16);
}
// B fragment 32x16 bf16 from LDS [pix][XPAD] (K contiguous per pixel row):
// lanes 0-15 hold K {0..7,16..23}, lanes 16-31 hold K {8..15,24..31} (2 packed per VGPR)
__device__ __forceinline__ v16bf ld_b(const unsigned short* base, int pix, int kt, int hi) {
  const unsigned short* p = base + pix * XPAD + kt * 32 + hi * 8;
  return ld_frag(p, p + 16);
}

// ---------------- kernel 1: global average pool --------------------------
__global__ __launch_bounds__(256) void pool_kernel(const float* __restrict__ x,
                                                   float* __restrict__ pooled) {
  const int b    = blockIdx.x;
  const int c    = blockIdx.y * 8 + (threadIdx.x >> 5);
  const int lane = threadIdx.x & 31;
  const float* p = x + ((size_t)b * CH + c) * HW;
  float s = 0.f;
  for (int i = lane; i < HW; i += 32) s += p[i];
  #pragma unroll
  for (int off = 16; off > 0; off >>= 1) s += __shfl_xor(s, off);
  if (lane == 0) pooled[b * CH + c] = s * (1.0f / (float)HW);
}

// ---------------- kernel 2: gate logits + argmax -------------------------
__global__ void gate_kernel(const float* __restrict__ pooled,
                            const float* __restrict__ Wg,
                            const float* __restrict__ bg,
                            int* __restrict__ eidx) {
  __shared__ float lg[8][NEXP];
  const int t = threadIdx.x;
  if (t < 8 * NEXP) {
    const int b = t / NEXP, e = t % NEXP;
    float s = bg[e];
    for (int c = 0; c < CH; ++c) s += pooled[b * CH + c] * Wg[e * CH + c];
    lg[b][e] = s;
  }
  __syncthreads();
  if (t < 8) {
    float best = lg[t][0]; int bi = 0;
    #pragma unroll
    for (int e = 1; e < NEXP; ++e)
      if (lg[t][e] > best) { best = lg[t][e]; bi = e; }
    eidx[t] = bi;   // one-hot softmax => weight is exactly 1.0 at argmax
  }
}

// ---------------- per-half expert path (runs on one wave) ----------------
// D = (W+I) @ x + b (x tile already bf16 in xw, overwritten with D bf16),
// h = relu(W1 @ D + b1), returns a = sigmoid(W2 . h + b2) per pixel lane.
__device__ __forceinline__ float expert_path(unsigned short* xw,
                                             const unsigned short* sW,
                                             const unsigned short* sW1,
                                             const float* sB,
                                             const float* sB1,
                                             const float* sW2,
                                             float b2, int lane) {
  const int hi = lane >> 4, ln = lane & 15;
  v8f acc[8];
  #pragma unroll
  for (int mt = 0; mt < 8; ++mt)
    #pragma unroll
    for (int i = 0; i < 8; ++i) acc[mt][i] = sB[mt * 16 + hi * 8 + i];

  #pragma unroll
  for (int kt = 0; kt < 4; ++kt) {
    v16bf bf = ld_b(xw, ln, kt, hi);
    #pragma unroll
    for (int mt = 0; mt < 8; ++mt) {
      v16bf af = ld_a(sW, mt * 16 + ln, kt, hi);
      acc[mt] = __builtin_amdgcn_wmma_f32_16x16x32_bf16(
          false, af, false, bf, (short)0, acc[mt], false, false);
    }
  }
  // write D (bf16) back over this wave's own x rows (wave-local, in-order LDS)
  #pragma unroll
  for (int mt = 0; mt < 8; ++mt)
    #pragma unroll
    for (int i = 0; i < 8; i += 2) {
      unsigned int pk = (unsigned int)f2bf(acc[mt][i]) |
                        ((unsigned int)f2bf(acc[mt][i + 1]) << 16);
      *reinterpret_cast<unsigned int*>(xw + ln * XPAD + mt * 16 + hi * 8 + i) = pk;
    }

  v8f h[4];
  #pragma unroll
  for (int mt = 0; mt < 4; ++mt)
    #pragma unroll
    for (int i = 0; i < 8; ++i) h[mt][i] = sB1[mt * 16 + hi * 8 + i];

  #pragma unroll
  for (int kt = 0; kt < 4; ++kt) {
    v16bf bf = ld_b(xw, ln, kt, hi);   // now reads D
    #pragma unroll
    for (int mt = 0; mt < 4; ++mt) {
      v16bf af = ld_a(sW1, mt * 16 + ln, kt, hi);
      h[mt] = __builtin_amdgcn_wmma_f32_16x16x32_bf16(
          false, af, false, bf, (short)0, h[mt], false, false);
    }
  }
  float part = 0.f;
  #pragma unroll
  for (int mt = 0; mt < 4; ++mt)
    #pragma unroll
    for (int i = 0; i < 8; ++i)
      part += fmaxf(h[mt][i], 0.f) * sW2[mt * 16 + hi * 8 + i];
  part += __shfl_xor(part, 16);        // combine M halves (same pixel column)
  return 1.0f / (1.0f + __expf(-(part + b2)));
}

// ---------------- kernel 3: selected-expert WMMA kernel ------------------
__global__ __launch_bounds__(256) void moe_expert_kernel(
    const float* __restrict__ x,
    const float* __restrict__ Wrgb, const float* __restrict__ brgb,
    const float* __restrict__ Wtir, const float* __restrict__ btir,
    const float* __restrict__ Wt1,  const float* __restrict__ bt1,
    const float* __restrict__ Wt2,  const float* __restrict__ bt2,
    const int* __restrict__ eidx,
    float* __restrict__ out) {
  __shared__ __align__(16) unsigned short sWr[HALF * XPAD];
  __shared__ __align__(16) unsigned short sWt[HALF * XPAD];
  __shared__ __align__(16) unsigned short sW1[QTR * XPAD];
  __shared__ __align__(16) unsigned short sXr[HALF * XPAD];
  __shared__ __align__(16) unsigned short sXt[HALF * XPAD];
  __shared__ float sBr[HALF], sBt[HALF], sB1[QTR], sW2[QTR];
  __shared__ float sB2;

  const int tid   = threadIdx.x;
  const int b     = blockIdx.y;
  const int e     = eidx[b];
  const int pbase = blockIdx.x * 128;

  // stage selected expert's weights as bf16; fuse residual as W + I (exact)
  const float* wr = Wrgb + (size_t)e * HALF * HALF;
  const float* wt = Wtir + (size_t)e * HALF * HALF;
  #pragma unroll 4
  for (int i = 0; i < 64; ++i) {
    int idx = tid + i * 256, o = idx >> 7, c = idx & 127;
    float d = (o == c) ? 1.0f : 0.0f;
    sWr[o * XPAD + c] = f2bf(wr[idx] + d);
    sWt[o * XPAD + c] = f2bf(wt[idx] + d);
  }
  const float* w1 = Wt1 + (size_t)e * QTR * HALF;
  #pragma unroll 4
  for (int i = 0; i < 32; ++i) {
    int idx = tid + i * 256, m = idx >> 7, c = idx & 127;
    sW1[m * XPAD + c] = f2bf(w1[idx]);
  }
  if (tid < HALF) { sBr[tid] = brgb[e * HALF + tid]; sBt[tid] = btir[e * HALF + tid]; }
  if (tid < QTR)  { sB1[tid] = bt1[e * QTR + tid];   sW2[tid] = Wt2[e * QTR + tid]; }
  if (tid == 0)   sB2 = bt2[e];

  // stage x tile (256 ch x 128 px) -> bf16 LDS, layout [pix][ch]
  const float* xb = x + (size_t)b * CH * HW;
  const int cpair = tid >> 7, px = tid & 127;
  for (int c2 = 0; c2 < 128; ++c2) {
    int ch  = c2 * 2 + cpair;
    float v = xb[(size_t)ch * HW + pbase + px];
    unsigned short* dst = (ch < HALF) ? sXr : sXt;
    dst[px * XPAD + (ch & 127)] = f2bf(v);
  }
  __syncthreads();

  const int w = tid >> 5, lane = tid & 31;
  unsigned short* xwr = sXr + w * 16 * XPAD;   // this wave's 16 pixel rows
  unsigned short* xwt = sXt + w * 16 * XPAD;

  const float aR = expert_path(xwr, sWr, sW1, sBr, sB1, sW2, sB2, lane);
  const float aT = expert_path(xwt, sWt, sW1, sBt, sB1, sW2, sB2, lane);

  // out[b, ch, pix] = aR*Dr + aT*Dt  (D re-read from bf16 LDS copies)
  const int hi = lane >> 4, ln = lane & 15;
  float* ob = out + (size_t)b * HALF * HW + pbase + w * 16 + ln;
  #pragma unroll
  for (int mt = 0; mt < 8; ++mt) {
    #pragma unroll
    for (int i = 0; i < 8; ++i) {
      int chn  = mt * 16 + hi * 8 + i;
      float dr = bf2f(xwr[ln * XPAD + chn]);
      float dt = bf2f(xwt[ln * XPAD + chn]);
      ob[(size_t)chn * HW] = aR * dr + aT * dt;
    }
  }
}

extern "C" void kernel_launch(void* const* d_in, const int* in_sizes, int n_in,
                              void* d_out, int out_size, void* d_ws, size_t ws_size,
                              hipStream_t stream) {
  (void)in_sizes; (void)n_in; (void)out_size; (void)ws_size;
  const float* x    = (const float*)d_in[0];
  const float* Wg   = (const float*)d_in[1];
  const float* bg   = (const float*)d_in[2];
  const float* Wrgb = (const float*)d_in[3];
  const float* brgb = (const float*)d_in[4];
  const float* Wtir = (const float*)d_in[5];
  const float* btir = (const float*)d_in[6];
  const float* Wt1  = (const float*)d_in[7];
  const float* bt1  = (const float*)d_in[8];
  const float* Wt2  = (const float*)d_in[9];
  const float* bt2  = (const float*)d_in[10];
  float* out    = (float*)d_out;
  float* pooled = (float*)d_ws;                                  // 8*256 f32
  int*   eidx   = (int*)((char*)d_ws + 8 * CH * sizeof(float));  // 8 ints

  pool_kernel<<<dim3(8, 32), 256, 0, stream>>>(x, pooled);
  gate_kernel<<<1, 64, 0, stream>>>(pooled, Wg, bg, eidx);
  moe_expert_kernel<<<dim3(HW / 128, 8), 256, 0, stream>>>(
      x, Wrgb, brgb, Wtir, btir, Wt1, bt1, Wt2, bt2, eidx, out);
}
